// DoubleALIFRNN_52759378264214
// MI455X (gfx1250) — matrix operand order repaired and checked by reading, not compiled, not run on hardware
//
#include <hip/hip_runtime.h>
#include <stdint.h>

// ---------------- problem constants ----------------
#define S_T  1000
#define BB   256
#define II   512
#define HH   1024
#define OO   128
#define NWG  64
#define TPB  256
#define LDST 40        // LDS row stride in ushorts (32 + 8 pad, spreads banks)

// ---------------- workspace layout (bytes) ----------------
constexpr size_t OFF_BAR = 0;                                  // 2 x u32 barrier state
constexpr size_t OFF_W1  = 256;
constexpr size_t SZ_W1   = (size_t)HH * (II + HH) * 2;         // bf16
constexpr size_t OFF_W2  = OFF_W1 + SZ_W1;
constexpr size_t SZ_W2   = (size_t)HH * (2 * HH) * 2;
constexpr size_t OFF_WO  = OFF_W2 + SZ_W2;
constexpr size_t SZ_WO   = (size_t)OO * HH * 2;
constexpr size_t OFF_Z1  = OFF_WO + SZ_WO;                     // 2 parities, bf16
constexpr size_t SZ_Z    = 2ull * BB * HH * 2;
constexpr size_t OFF_Z2  = OFF_Z1 + SZ_Z;
constexpr size_t OFF_U1  = OFF_Z2 + SZ_Z;                      // f32 states
constexpr size_t SZ_ST   = (size_t)BB * HH * 4;
constexpr size_t OFF_A1  = OFF_U1 + SZ_ST;
constexpr size_t OFF_U2  = OFF_A1 + SZ_ST;
constexpr size_t OFF_A2  = OFF_U2 + SZ_ST;
constexpr size_t OFF_UO  = OFF_A2 + SZ_ST;
constexpr size_t SZ_UO   = (size_t)BB * OO * 4;
constexpr size_t OFF_AL1 = OFF_UO + SZ_UO;                     // per-neuron decay consts
constexpr size_t OFF_RH1 = OFF_AL1 + 4096;
constexpr size_t OFF_AL2 = OFF_RH1 + 4096;
constexpr size_t OFF_RH2 = OFF_AL2 + 4096;
constexpr size_t OFF_ALO = OFF_RH2 + 4096;
constexpr size_t ZERO_W  = (OFF_AL1 - OFF_Z1) / 4;             // words to zero (z/u/a/uo)

// ---------------- types ----------------
typedef __attribute__((ext_vector_type(16))) __bf16         v16bf;
typedef __attribute__((ext_vector_type(8)))  float          v8f;
typedef __attribute__((ext_vector_type(8)))  unsigned short v8u;
typedef __attribute__((ext_vector_type(16))) unsigned short v16u;

union V16 { v16u u; v16bf b; };

__device__ __forceinline__ unsigned short f2bf(float f) {
  unsigned u = __float_as_uint(f);
  u += 0x7FFFu + ((u >> 16) & 1u);      // round-to-nearest-even
  return (unsigned short)(u >> 16);
}

// ---------------- CDNA5 async global->LDS copy (ASYNCcnt path) ----------------
// ISA 15.18.3 op 98: GLOBAL_LOAD_ASYNC_TO_LDS_B128, VDST = LDS byte address VGPR.
// For a __shared__ object, the low 32 bits of the flat address ARE the LDS address
// (ISA 10.2).
__device__ __forceinline__ void asyncCp16(unsigned short* ldsDst, const unsigned short* gsrc) {
  unsigned off = (unsigned)(size_t)ldsDst;
  asm volatile("global_load_async_to_lds_b128 %0, %1, off"
               :: "v"(off), "v"(gsrc)
               : "memory");
}
__device__ __forceinline__ void waitAsync() {
  asm volatile("s_wait_asynccnt 0x0" ::: "memory");
}

// ---------------- WMMA fragment loads ----------------
// A (16x32 bf16): lane<16 -> row=lane, K: k0..k0+7 (v0-3) and k0+16..k0+23 (v4-7), k0=0
//                 lane>=16 -> same rows, k0=8.
__device__ __forceinline__ v16bf ldsA(const unsigned short* base, int mSub, int lane) {
  const unsigned short* p = base + (size_t)(mSub * 16 + (lane & 15)) * LDST + ((lane >> 4) << 3);
  v8u lo = *(const v8u*)p;
  v8u hi = *(const v8u*)(p + 16);
  V16 r;
#pragma unroll
  for (int i = 0; i < 8; ++i) { r.u[i] = lo[i]; r.u[i + 8] = hi[i]; }
  return r.b;
}
// B (32x16 bf16, stored n-major rows of k): lane<16 -> col n=lane, K=0..7,16..23; lane>=16 -> K=8..15,24..31.
__device__ __forceinline__ v16bf ldsB(const unsigned short* base, int nSub, int lane) {
  const unsigned short* p = base + (size_t)(nSub * 16 + (lane & 15)) * LDST + ((lane >> 4) << 3);
  v8u lo = *(const v8u*)p;
  v8u hi = *(const v8u*)(p + 16);
  V16 r;
#pragma unroll
  for (int i = 0; i < 8; ++i) { r.u[i] = lo[i]; r.u[i + 8] = hi[i]; }
  return r.b;
}
// same patterns straight from global (row stride HH) for the small readout GEMM
__device__ __forceinline__ v16bf glbA(const unsigned short* base, int lane) {
  const unsigned short* p = base + (size_t)(lane & 15) * HH + ((lane >> 4) << 3);
  v8u lo = *(const v8u*)p;
  v8u hi = *(const v8u*)(p + 16);
  V16 r;
#pragma unroll
  for (int i = 0; i < 8; ++i) { r.u[i] = lo[i]; r.u[i + 8] = hi[i]; }
  return r.b;
}
__device__ __forceinline__ v16bf glbB(const unsigned short* base, int lane) {
  const unsigned short* p = base + (size_t)(lane & 15) * HH + ((lane >> 4) << 3);
  v8u lo = *(const v8u*)p;
  v8u hi = *(const v8u*)(p + 16);
  V16 r;
#pragma unroll
  for (int i = 0; i < 8; ++i) { r.u[i] = lo[i]; r.u[i + 8] = hi[i]; }
  return r.b;
}

#define WMMA_BF16(A, B, C) \
  __builtin_amdgcn_wmma_f32_16x16x32_bf16(false, (A), false, (B), (short)0, (C), false, false)

// ---------------- device-wide sense barrier ----------------
__device__ __forceinline__ void gridBarrier(unsigned* bar) {
  __threadfence();
  __syncthreads();
  if (threadIdx.x == 0) {
    unsigned g = __hip_atomic_load(&bar[1], __ATOMIC_RELAXED, __HIP_MEMORY_SCOPE_AGENT);
    unsigned prev = __hip_atomic_fetch_add(&bar[0], 1u, __ATOMIC_ACQ_REL, __HIP_MEMORY_SCOPE_AGENT);
    if (prev == NWG - 1) {
      __hip_atomic_store(&bar[0], 0u, __ATOMIC_RELAXED, __HIP_MEMORY_SCOPE_AGENT);
      __hip_atomic_fetch_add(&bar[1], 1u, __ATOMIC_RELEASE, __HIP_MEMORY_SCOPE_AGENT);
    } else {
      while (__hip_atomic_load(&bar[1], __ATOMIC_ACQUIRE, __HIP_MEMORY_SCOPE_AGENT) == g) {
        __builtin_amdgcn_s_sleep(2);
      }
    }
  }
  __syncthreads();
  __threadfence();
}

// ---------------- fused ALIF epilogue on a 16x16 accumulator tile ----------------
__device__ __forceinline__ void alifUpd(const v8f& acc, int mB, int nB,
                                        float* __restrict__ U, float* __restrict__ A,
                                        const unsigned short* __restrict__ zr,
                                        unsigned short* __restrict__ zw,
                                        const float* __restrict__ alpha,
                                        const float* __restrict__ rho, int lane) {
  const int n  = nB + (lane & 15);
  const int m0 = mB + ((lane >> 4) << 3);
  const float al = alpha[n], rh = rho[n];
#pragma unroll
  for (int r = 0; r < 8; ++r) {
    size_t idx = (size_t)(m0 + r) * HH + n;
    float u = U[idx];
    float a = A[idx];
    float z = zr[idx] ? 1.0f : 0.0f;            // spikes stored as bf16 {0,1}
    a = rh * a + (1.0f - rh) * z;
    float th = 0.01f + 1.8f * a;                // B0 + BETA*a
    u = al * u + (1.0f - al) * acc[r] - th * z; // soft reset
    U[idx] = u;
    A[idx] = a;
    zw[idx] = (u - th) > 0.0f ? (unsigned short)0x3F80 : (unsigned short)0; // bf16 1.0/0.0
  }
}

// ---------------- init: convert weights to bf16, precompute decays, zero state ----------------
__global__ __launch_bounds__(TPB) void alif_init(
    const float* __restrict__ W1, const float* __restrict__ W2, const float* __restrict__ Wo,
    const float* __restrict__ tm1, const float* __restrict__ ta1,
    const float* __restrict__ tm2, const float* __restrict__ ta2,
    const float* __restrict__ to, char* __restrict__ ws) {
  size_t i  = (size_t)blockIdx.x * blockDim.x + threadIdx.x;
  size_t st = (size_t)gridDim.x * blockDim.x;
  unsigned short* w1b = (unsigned short*)(ws + OFF_W1);
  unsigned short* w2b = (unsigned short*)(ws + OFF_W2);
  unsigned short* wob = (unsigned short*)(ws + OFF_WO);
  for (size_t k = i; k < (size_t)HH * (II + HH); k += st) w1b[k] = f2bf(W1[k]);
  for (size_t k = i; k < (size_t)HH * (2 * HH);  k += st) w2b[k] = f2bf(W2[k]);
  for (size_t k = i; k < (size_t)OO * HH;        k += st) wob[k] = f2bf(Wo[k]);
  unsigned* zr = (unsigned*)(ws + OFF_Z1);
  for (size_t k = i; k < ZERO_W; k += st) zr[k] = 0u;          // z1,z2,u,a,uo = 0
  float* al1 = (float*)(ws + OFF_AL1);
  float* rh1 = (float*)(ws + OFF_RH1);
  float* al2 = (float*)(ws + OFF_AL2);
  float* rh2 = (float*)(ws + OFF_RH2);
  float* alo = (float*)(ws + OFF_ALO);
  for (size_t k = i; k < HH; k += st) {
    al1[k] = expf(-1.0f / fabsf(tm1[k]));
    rh1[k] = expf(-1.0f / fabsf(ta1[k]));
    al2[k] = expf(-1.0f / fabsf(tm2[k]));
    rh2[k] = expf(-1.0f / fabsf(ta2[k]));
  }
  for (size_t k = i; k < OO; k += st) alo[k] = expf(-1.0f / fabsf(to[k]));
  if (i < 64) ((unsigned*)(ws + OFF_BAR))[i] = 0u;             // barrier arrive/gen = 0
}

// ---------------- persistent time-loop kernel ----------------
__global__ __launch_bounds__(TPB) void alif_persist(
    const float* __restrict__ x, float* __restrict__ out, char* __restrict__ ws) {
  const int tid  = threadIdx.x;
  const int lane = tid & 31;
  const int wave = tid >> 5;                 // 0..7
  const int wg   = blockIdx.x;               // 0..63
  const int mBase = (wg & 3) * 64;           // M block (256/64 = 4)
  const int nBase = (wg >> 2) * 64;          // N block (1024/64 = 16)
  const int mSub  = wave & 3;                // 16-row subtile
  const int nPair = wave >> 2;               // owns n-subtiles {2p, 2p+1}
  const int sRow  = tid >> 2;                // staging: row 0..63
  const int sCol  = (tid & 3) << 3;          // staging: 8-elem column group

  unsigned short* W1b = (unsigned short*)(ws + OFF_W1);
  unsigned short* W2b = (unsigned short*)(ws + OFF_W2);
  unsigned short* Wob = (unsigned short*)(ws + OFF_WO);
  unsigned short* Z1  = (unsigned short*)(ws + OFF_Z1);
  unsigned short* Z2  = (unsigned short*)(ws + OFF_Z2);
  float* U1 = (float*)(ws + OFF_U1);
  float* A1 = (float*)(ws + OFF_A1);
  float* U2 = (float*)(ws + OFF_U2);
  float* A2 = (float*)(ws + OFF_A2);
  float* UO = (float*)(ws + OFF_UO);
  const float* al1 = (const float*)(ws + OFF_AL1);
  const float* rh1 = (const float*)(ws + OFF_RH1);
  const float* al2 = (const float*)(ws + OFF_AL2);
  const float* rh2 = (const float*)(ws + OFF_RH2);
  const float* alo = (const float*)(ws + OFF_ALO);
  unsigned* bar = (unsigned*)(ws + OFF_BAR);

  __shared__ unsigned short As[2][64 * LDST];   // double-buffered slabs
  __shared__ unsigned short Bs[2][64 * LDST];

  const size_t ldsIdx = (size_t)sRow * LDST + sCol;

  for (int t = 0; t < S_T; ++t) {
    const int p = t & 1, q = p ^ 1;
    const unsigned short* z1r = Z1 + (size_t)p * (BB * HH);
    unsigned short*       z1w = Z1 + (size_t)q * (BB * HH);
    const unsigned short* z2r = Z2 + (size_t)p * (BB * HH);
    unsigned short*       z2w = Z2 + (size_t)q * (BB * HH);
    const float* xt = x + (size_t)t * (BB * II);

    // ===== layer 1: cur = [x_t | z1_{t-1}] @ W1^T, fused ALIF =====
    {
      const int m = mBase + sRow;
      const int n = nBase + sRow;
      v8f acc0 = {}, acc1 = {};

      // stage(i): slab kb = 32*i into buffer buf. x part converts via VALU+ds_store,
      // bf16 parts and weights go through the async global->LDS pipe.
      auto stage1 = [&](int i, int buf) {
        const int kb = i * 32;
        unsigned short* dstA = &As[buf][ldsIdx];
        if (kb < II) {
          const float* sp = xt + (size_t)m * II + kb + sCol;
          const float4 f0 = ((const float4*)sp)[0];
          const float4 f1 = ((const float4*)sp)[1];
          v8u v;
          v[0] = f2bf(f0.x); v[1] = f2bf(f0.y); v[2] = f2bf(f0.z); v[3] = f2bf(f0.w);
          v[4] = f2bf(f1.x); v[5] = f2bf(f1.y); v[6] = f2bf(f1.z); v[7] = f2bf(f1.w);
          *(v8u*)dstA = v;
        } else {
          asyncCp16(dstA, z1r + (size_t)m * HH + (kb - II) + sCol);
        }
        asyncCp16(&Bs[buf][ldsIdx], W1b + (size_t)n * (II + HH) + kb + sCol);
      };

      constexpr int NK1 = (II + HH) / 32;  // 48
      stage1(0, 0);
      for (int i = 0; i < NK1; ++i) {
        const int cur = i & 1;
        waitAsync();            // our writes for buf[cur] have landed in LDS
        __syncthreads();        // publish buf[cur]; everyone done reading buf[cur^1]
        if (i + 1 < NK1) stage1(i + 1, cur ^ 1);  // overlap with WMMA below
        v16bf a  = ldsA(As[cur], mSub, lane);
        v16bf b0 = ldsB(Bs[cur], nPair * 2,     lane);
        v16bf b1 = ldsB(Bs[cur], nPair * 2 + 1, lane);
        acc0 = WMMA_BF16(a, b0, acc0);
        acc1 = WMMA_BF16(a, b1, acc1);
      }
      alifUpd(acc0, mBase + mSub * 16, nBase + nPair * 32,      U1, A1, z1r, z1w, al1, rh1, lane);
      alifUpd(acc1, mBase + mSub * 16, nBase + nPair * 32 + 16, U1, A1, z1r, z1w, al1, rh1, lane);
    }
    gridBarrier(bar);   // z1_t visible device-wide

    // ===== layer 2: cur = [z1_t | z2_{t-1}] @ W2^T, fused ALIF =====
    {
      const int m = mBase + sRow;
      const int n = nBase + sRow;
      v8f acc0 = {}, acc1 = {};

      auto stage2 = [&](int i, int buf) {
        const int kb = i * 32;
        const unsigned short* src = (kb < HH)
            ? (z1w + (size_t)m * HH + kb + sCol)
            : (z2r + (size_t)m * HH + (kb - HH) + sCol);
        asyncCp16(&As[buf][ldsIdx], src);
        asyncCp16(&Bs[buf][ldsIdx], W2b + (size_t)n * (2 * HH) + kb + sCol);
      };

      constexpr int NK2 = (2 * HH) / 32;   // 64
      stage2(0, 0);
      for (int i = 0; i < NK2; ++i) {
        const int cur = i & 1;
        waitAsync();
        __syncthreads();
        if (i + 1 < NK2) stage2(i + 1, cur ^ 1);
        v16bf a  = ldsA(As[cur], mSub, lane);
        v16bf b0 = ldsB(Bs[cur], nPair * 2,     lane);
        v16bf b1 = ldsB(Bs[cur], nPair * 2 + 1, lane);
        acc0 = WMMA_BF16(a, b0, acc0);
        acc1 = WMMA_BF16(a, b1, acc1);
      }
      alifUpd(acc0, mBase + mSub * 16, nBase + nPair * 32,      U2, A2, z2r, z2w, al2, rh2, lane);
      alifUpd(acc1, mBase + mSub * 16, nBase + nPair * 32 + 16, U2, A2, z2r, z2w, al2, rh2, lane);
    }
    gridBarrier(bar);   // z2_t visible device-wide

    // ===== readout: uo = a_o*uo + (1-a_o)*(z2_t @ Wout^T), write out[t] =====
    {
      const int wgid = wg * 8 + wave;                    // 0..511
      if (wgid < (BB / 16) * (OO / 16)) {                // 128 waves, one 16x16 tile each
        const int mT = wgid & 15, nT = wgid >> 4;
        v8f acc = {};
        for (int kb = 0; kb < HH; kb += 32) {
          v16bf a = glbA(z2w + (size_t)(mT * 16) * HH + kb, lane);
          v16bf b = glbB(Wob + (size_t)(nT * 16) * HH + kb, lane);
          acc = WMMA_BF16(a, b, acc);
        }
        const int n  = nT * 16 + (lane & 15);
        const int m0 = mT * 16 + ((lane >> 4) << 3);
        const float al = alo[n];
#pragma unroll
        for (int r = 0; r < 8; ++r) {
          size_t idx = (size_t)(m0 + r) * OO + n;
          float v = al * UO[idx] + (1.0f - al) * acc[r];
          UO[idx] = v;
          out[(size_t)t * (BB * OO) + idx] = v;
        }
      }
    }
    // next step's writes land in the opposite parity buffers; the two barriers
    // above already separate them from this step's readers.
  }
}

extern "C" void kernel_launch(void* const* d_in, const int* in_sizes, int n_in,
                              void* d_out, int out_size, void* d_ws, size_t ws_size,
                              hipStream_t stream) {
  const float* x   = (const float*)d_in[0];
  const float* W1  = (const float*)d_in[1];
  const float* W2  = (const float*)d_in[2];
  const float* Wo  = (const float*)d_in[3];
  const float* tm1 = (const float*)d_in[4];
  const float* ta1 = (const float*)d_in[5];
  const float* tm2 = (const float*)d_in[6];
  const float* ta2 = (const float*)d_in[7];
  const float* to  = (const float*)d_in[8];
  char* ws = (char*)d_ws;   // requires ~13.4 MB of workspace

  alif_init<<<1024, TPB, 0, stream>>>(W1, W2, Wo, tm1, ta1, tm2, ta2, to, ws);
  alif_persist<<<NWG, TPB, 0, stream>>>(x, (float*)d_out, ws);
  (void)in_sizes; (void)n_in; (void)out_size; (void)ws_size;
}
